// smallest_k_dist_loss_52664888984155
// MI455X (gfx1250) — matrix operand order repaired and checked by reading, not compiled, not run on hardware
//
#include <hip/hip_runtime.h>
#include <math.h>

// ---------------------------------------------------------------------------
// Border-distance loss for a 3-layer MLP, CDNA5 (gfx1250, wave32, WMMA).
// B=256, D=256, d1=1024, d2=512, d3=256, k read from device scalar.
//
// All GEMMs use v_wmma_f32_16x16x32_f16 with split-f32 (hi/lo f16) operands
// giving ~f32 accuracy at f16 matrix throughput. The hi/lo planes are
// PRE-SPLIT in workspace memory so inner loops are pure b128 loads + WMMA
// (no cvt chains) -> matrix pipe bound, not VALU bound. The whole working
// set (~10 MB) is L2-resident on MI455X (192 MB L2), so this is compute-
// bound; the ReLU-mask is applied as an exact f16 multiply by {1,0}.
// ---------------------------------------------------------------------------

typedef __attribute__((ext_vector_type(16))) _Float16 v16h;
typedef __attribute__((ext_vector_type(8)))  _Float16 v8h;
typedef __attribute__((ext_vector_type(8)))  float    v8f;

// ---- WMMA fragment loaders (contiguous k = 16h + s, same for A and B, wave32) ----------
// A (16x32 f16, row-major source MxK):
//   lanes 0-15: M=lane,  elems 0..15 = K k0+0..15
//   lanes 16-31: M=lane-16, elems 0..15 = K k0+16..31
__device__ inline v16h load_frag_a(const _Float16* __restrict__ P, int lda,
                                   int row0, int k0, int lane) {
  const int m  = row0 + (lane & 15);
  const int kb = k0 + ((lane >> 4) << 4);
  const _Float16* p = P + m * lda + kb;
  const v8h x0 = *(const v8h*)(p);        // 16B aligned
  const v8h x1 = *(const v8h*)(p + 8);    // 16B aligned
  return __builtin_shufflevector(x0, x1, 0, 1, 2, 3, 4, 5, 6, 7,
                                 8, 9, 10, 11, 12, 13, 14, 15);
}

// B (32x16 f16) from BT stored [N x K] row-major (BT[n][k] = B[k][n]):
//   lanes 0-15: N=lane, elems 0..15 = K k0+0..15 ; lanes 16-31: K k0+16..31
__device__ inline v16h load_frag_b(const _Float16* __restrict__ P, int ldb,
                                   int col0, int k0, int lane) {
  const int n  = col0 + (lane & 15);
  const int kb = k0 + ((lane >> 4) << 4);
  const _Float16* p = P + n * ldb + kb;
  const v8h x0 = *(const v8h*)(p);
  const v8h x1 = *(const v8h*)(p + 8);
  return __builtin_shufflevector(x0, x1, 0, 1, 2, 3, 4, 5, 6, 7,
                                 8, 9, 10, 11, 12, 13, 14, 15);
}

// Per-K mask fragment ({1.0h, 0.0h} vector, broadcast over N lanes).
__device__ inline v16h load_frag_mask(const _Float16* M, int k0, int lane) {
  const int kb = k0 + ((lane >> 4) << 4);
  const v8h x0 = *(const v8h*)(M + kb);
  const v8h x1 = *(const v8h*)(M + kb + 8);
  return __builtin_shufflevector(x0, x1, 0, 1, 2, 3, 4, 5, 6, 7,
                                 8, 9, 10, 11, 12, 13, 14, 15);
}

// (ahi+alo)*(bhi+blo) ~= a*b, dropping the O(2^-22) lo*lo term.
__device__ inline v8f wmma3(v16h ah, v16h al, v16h bh, v16h bl, v8f c) {
  c = __builtin_amdgcn_wmma_f32_16x16x32_f16(false, al, false, bh, (short)0, c, false, false);
  c = __builtin_amdgcn_wmma_f32_16x16x32_f16(false, ah, false, bl, (short)0, c, false, false);
  c = __builtin_amdgcn_wmma_f32_16x16x32_f16(false, ah, false, bh, (short)0, c, false, false);
  return c;
}

// ---- prep kernels ----------------------------------------------------------
__global__ void cvt_split(const float* __restrict__ src,
                          _Float16* __restrict__ hi, _Float16* __restrict__ lo,
                          int n) {
  for (int i = blockIdx.x * blockDim.x + threadIdx.x; i < n;
       i += gridDim.x * blockDim.x) {
    const float v = src[i];
    const _Float16 h = (_Float16)v;
    hi[i] = h;
    lo[i] = (_Float16)(v - (float)h);
  }
}

__global__ void transpose_split(const float* __restrict__ in,
                                _Float16* __restrict__ thi,
                                _Float16* __restrict__ tlo, int R, int C) {
  const int total = R * C;
  for (int i = blockIdx.x * blockDim.x + threadIdx.x; i < total;
       i += gridDim.x * blockDim.x) {
    const int r = i / C, c = i - r * C;
    const float v = in[i];
    const _Float16 h = (_Float16)v;
    thi[c * R + r] = h;
    tlo[c * R + r] = (_Float16)(v - (float)h);
  }
}

__global__ void rownorm(const float* __restrict__ W, float* __restrict__ norms,
                        int cols) {
  __shared__ float red[256];
  const int row = blockIdx.x;
  float s = 0.0f;
  for (int c = threadIdx.x; c < cols; c += blockDim.x) {
    const float v = W[row * cols + c];
    s += v * v;
  }
  red[threadIdx.x] = s;
  __syncthreads();
  for (int st = 128; st > 0; st >>= 1) {
    if (threadIdx.x < st) red[threadIdx.x] += red[threadIdx.x + st];
    __syncthreads();
  }
  if (threadIdx.x == 0) norms[row] = sqrtf(red[0]);
}

// ---- generic WMMA GEMM: Out[MxN] = A[MxK]*B[KxN] + bias, operands pre-split
__global__ void __launch_bounds__(256) gemm_bias_store(
    const _Float16* __restrict__ Ah, const _Float16* __restrict__ Al,
    const _Float16* __restrict__ Bh, const _Float16* __restrict__ Bl,
    const float* __restrict__ bias, float* __restrict__ Out,
    int M, int N, int K) {
  const int lane = threadIdx.x & 31;
  const int wave = threadIdx.x >> 5;
  const int wpb  = blockDim.x >> 5;
  const int nt   = N >> 4;
  const int tiles = (M >> 4) * nt;
  for (int t = blockIdx.x * wpb + wave; t < tiles; t += gridDim.x * wpb) {
    const int m0 = (t / nt) << 4;
    const int n0 = (t % nt) << 4;
    v8f c = {0.f, 0.f, 0.f, 0.f, 0.f, 0.f, 0.f, 0.f};
    for (int k0 = 0; k0 < K; k0 += 32) {
      const v16h ah = load_frag_a(Ah, K, m0, k0, lane);
      const v16h al = load_frag_a(Al, K, m0, k0, lane);
      const v16h bh = load_frag_b(Bh, K, n0, k0, lane);
      const v16h bl = load_frag_b(Bl, K, n0, k0, lane);
      c = wmma3(ah, al, bh, bl, c);
    }
    // C/D layout: lanes 0-15 N=lane, M=m0+r ; lanes 16-31 N=lane-16, M=m0+8+r
    const int n  = n0 + (lane & 15);
    const int mb = m0 + ((lane >> 4) << 3);
    const float bn = bias[n];
#pragma unroll
    for (int r = 0; r < 8; ++r) Out[(mb + r) * N + n] = c[r] + bn;
  }
}

// ---- activation packing: relu -> hi/lo planes + sign bitmask (wave32) ------
__global__ void pack_layer1(const float* __restrict__ z1,
                            const float* __restrict__ w1norm,
                            _Float16* __restrict__ h1hi,
                            _Float16* __restrict__ h1lo,
                            unsigned* __restrict__ m1bits,
                            float* __restrict__ dists) {
  const int idx = blockIdx.x * blockDim.x + threadIdx.x;  // exactly 256*1024
  const int b = idx >> 10;
  const int j = idx & 1023;
  const float z = z1[idx];
  const float h = z > 0.0f ? z : 0.0f;
  const _Float16 hh = (_Float16)h;
  h1hi[idx] = hh;
  h1lo[idx] = (_Float16)(h - (float)hh);
  dists[b * 1792 + j] = fabsf(z) / w1norm[j];
  const unsigned long long bal = __ballot(z > 0.0f);
  if ((threadIdx.x & 31) == 0) m1bits[idx >> 5] = (unsigned)bal;
}

__global__ void pack_layer2(const float* __restrict__ z2,
                            _Float16* __restrict__ h2hi,
                            _Float16* __restrict__ h2lo,
                            unsigned* __restrict__ m2bits) {
  const int idx = blockIdx.x * blockDim.x + threadIdx.x;  // exactly 256*512
  const float z = z2[idx];
  const float h = z > 0.0f ? z : 0.0f;
  const _Float16 hh = (_Float16)h;
  h2hi[idx] = hh;
  h2lo[idx] = (_Float16)(h - (float)hh);
  const unsigned long long bal = __ballot(z > 0.0f);
  if ((threadIdx.x & 31) == 0) m2bits[idx >> 5] = (unsigned)bal;
}

// ---- epilogues for the norm GEMMs ------------------------------------------
__device__ inline void gemm1_epilogue(v8f c, int m0, int lane,
                                      const unsigned* __restrict__ mw2,
                                      float* norm2,
                                      _Float16* v2t_hi, _Float16* v2t_lo) {
  const int n  = lane & 15;
  const int mb = m0 + ((lane >> 4) << 3);
#pragma unroll
  for (int r = 0; r < 8; ++r) {
    const int o = mb + r;
    const float v = c[r];
    float s = v * v;  // reduce the 16 columns of this row
    s += __shfl_xor(s, 1, 16);
    s += __shfl_xor(s, 2, 16);
    s += __shfl_xor(s, 4, 16);
    s += __shfl_xor(s, 8, 16);
    if (n == 0) norm2[o] += s;  // rows disjoint across waves
    const float vm = ((mw2[o >> 5] >> (o & 31)) & 1u) ? v : 0.0f;
    const _Float16 vh = (_Float16)vm;
    v2t_hi[n * 528 + o] = vh;
    v2t_lo[n * 528 + o] = (_Float16)(vm - (float)vh);
  }
}

__device__ inline void gemm2_epilogue(v8f c, int m0, int lane, float* norm3) {
  const int mb = m0 + ((lane >> 4) << 3);
#pragma unroll
  for (int r = 0; r < 8; ++r) {
    const float v = c[r];
    float s = v * v;
    s += __shfl_xor(s, 1, 16);
    s += __shfl_xor(s, 2, 16);
    s += __shfl_xor(s, 4, 16);
    s += __shfl_xor(s, 8, 16);
    if ((lane & 15) == 0) norm3[mb + r] += s;
  }
}

// ---- main per-batch kernel: V2/V3 row norms + dist2/dist3 ------------------
// One workgroup (8 waves) per batch element. Loops D in 16-wide tiles:
//   V2tile[512x16] = W2 * (diag(m1) W1)[:,tile]   (WMMA, mask = f16 {1,0} mul)
//   norm2 += rowsum(V2tile^2); stage m2-masked V2tile in LDS (hi/lo planes)
//   V3tile[256x16] = W3 * V2tile                  (WMMA, B from LDS b128)
//   norm3 += rowsum(V3tile^2)
// Each wave holds 4 (GEMM1) / 2 (GEMM2) accumulator tiles so B fragments and
// the mask multiply are amortized over 12 / 6 WMMAs.
__global__ void __launch_bounds__(256) border_norms(
    const _Float16* __restrict__ W2h, const _Float16* __restrict__ W2l,   // [512 x 1024]
    const _Float16* __restrict__ W1Th, const _Float16* __restrict__ W1Tl, // [256 x 1024]
    const _Float16* __restrict__ W3h, const _Float16* __restrict__ W3l,   // [256 x 512]
    const unsigned* __restrict__ m1bits,  // [B x 32] words
    const unsigned* __restrict__ m2bits,  // [B x 16] words
    const float* __restrict__ z2,   // [B x 512]
    const float* __restrict__ z3,   // [B x 256]
    float* __restrict__ dists) {    // [B x 1792]
  __shared__ __align__(16) _Float16 mask_h[1024];
  __shared__ __align__(16) _Float16 v2t_hi[16 * 528];  // [dcol][o], ld=528
  __shared__ __align__(16) _Float16 v2t_lo[16 * 528];
  __shared__ float norm2[512];
  __shared__ float norm3[256];
  const int b    = blockIdx.x;
  const int lane = threadIdx.x & 31;
  const int wave = threadIdx.x >> 5;
  const unsigned* mw1 = m1bits + b * 32;
  const unsigned* mw2 = m2bits + b * 16;

  for (int i = threadIdx.x; i < 1024; i += 256)
    mask_h[i] = ((mw1[i >> 5] >> (i & 31)) & 1u) ? (_Float16)1.0f : (_Float16)0.0f;
  for (int i = threadIdx.x; i < 512; i += 256) norm2[i] = 0.0f;
  for (int i = threadIdx.x; i < 256; i += 256) norm3[i] = 0.0f;
  __syncthreads();

  const v8f Z = {0.f, 0.f, 0.f, 0.f, 0.f, 0.f, 0.f, 0.f};
  const int m0 = wave * 16;  // this wave's base tile row

  for (int d0 = 0; d0 < 256; d0 += 16) {
    // ---- GEMM1: V2 column tile; wave owns rows m0, m0+128, m0+256, m0+384 --
    {
      v8f c0 = Z, c1 = Z, c2 = Z, c3 = Z;
      for (int k0 = 0; k0 < 1024; k0 += 32) {
        const v16h mk = load_frag_mask(mask_h, k0, lane);
        const v16h bh = load_frag_b(W1Th, 1024, d0, k0, lane) * mk;
        const v16h bl = load_frag_b(W1Tl, 1024, d0, k0, lane) * mk;
        {
          const v16h ah = load_frag_a(W2h, 1024, m0, k0, lane);
          const v16h al = load_frag_a(W2l, 1024, m0, k0, lane);
          c0 = wmma3(ah, al, bh, bl, c0);
        }
        {
          const v16h ah = load_frag_a(W2h, 1024, m0 + 128, k0, lane);
          const v16h al = load_frag_a(W2l, 1024, m0 + 128, k0, lane);
          c1 = wmma3(ah, al, bh, bl, c1);
        }
        {
          const v16h ah = load_frag_a(W2h, 1024, m0 + 256, k0, lane);
          const v16h al = load_frag_a(W2l, 1024, m0 + 256, k0, lane);
          c2 = wmma3(ah, al, bh, bl, c2);
        }
        {
          const v16h ah = load_frag_a(W2h, 1024, m0 + 384, k0, lane);
          const v16h al = load_frag_a(W2l, 1024, m0 + 384, k0, lane);
          c3 = wmma3(ah, al, bh, bl, c3);
        }
      }
      gemm1_epilogue(c0, m0,       lane, mw2, norm2, v2t_hi, v2t_lo);
      gemm1_epilogue(c1, m0 + 128, lane, mw2, norm2, v2t_hi, v2t_lo);
      gemm1_epilogue(c2, m0 + 256, lane, mw2, norm2, v2t_hi, v2t_lo);
      gemm1_epilogue(c3, m0 + 384, lane, mw2, norm2, v2t_hi, v2t_lo);
    }
    __syncthreads();

    // ---- GEMM2: V3 column tile; wave owns rows m0, m0+128 ------------------
    {
      v8f e0 = Z, e1 = Z;
      for (int k0 = 0; k0 < 512; k0 += 32) {
        const v16h bh = load_frag_b(v2t_hi, 528, 0, k0, lane);  // LDS b128
        const v16h bl = load_frag_b(v2t_lo, 528, 0, k0, lane);
        {
          const v16h ah = load_frag_a(W3h, 512, m0, k0, lane);
          const v16h al = load_frag_a(W3l, 512, m0, k0, lane);
          e0 = wmma3(ah, al, bh, bl, e0);
        }
        {
          const v16h ah = load_frag_a(W3h, 512, m0 + 128, k0, lane);
          const v16h al = load_frag_a(W3l, 512, m0 + 128, k0, lane);
          e1 = wmma3(ah, al, bh, bl, e1);
        }
      }
      gemm2_epilogue(e0, m0,       lane, norm3);
      gemm2_epilogue(e1, m0 + 128, lane, norm3);
    }
    __syncthreads();
  }

  for (int o = threadIdx.x; o < 512; o += 256) {
    const float zz = z2[b * 512 + o];
    dists[b * 1792 + 1024 + o] = fabsf(zz) / sqrtf(norm2[o]);
  }
  for (int o = threadIdx.x; o < 256; o += 256) {
    const float zz = z3[b * 256 + o];
    dists[b * 1792 + 1536 + o] = fabsf(zz) / sqrtf(norm3[o]);
  }
}

// ---- k-smallest selection per batch element (duplicate-safe, deterministic)
__global__ void topk_partial(const float* __restrict__ dists,
                             const int* __restrict__ kptr,
                             float* __restrict__ partial) {
  __shared__ float sd[1792];
  __shared__ float rv[256];
  __shared__ int   ri[256];
  __shared__ float acc_s;
  const int b = blockIdx.x, tid = threadIdx.x;
  for (int i = tid; i < 1792; i += 256) sd[i] = dists[b * 1792 + i];
  const int kk = *kptr;
  if (tid == 0) acc_s = 0.0f;
  __syncthreads();
  for (int it = 0; it < kk; ++it) {
    float mn = 3.4e38f;
    int   mi = 0;
    for (int i = tid; i < 1792; i += 256) {
      const float v = sd[i];
      if (v < mn) { mn = v; mi = i; }
    }
    rv[tid] = mn; ri[tid] = mi;
    __syncthreads();
    for (int s = 128; s > 0; s >>= 1) {
      if (tid < s && rv[tid + s] < rv[tid]) { rv[tid] = rv[tid + s]; ri[tid] = ri[tid + s]; }
      __syncthreads();
    }
    if (tid == 0) { acc_s += rv[0]; sd[ri[0]] = 3.4e38f; }
    __syncthreads();
  }
  if (tid == 0) partial[b] = acc_s;
}

__global__ void final_sum(const float* __restrict__ partial, float* __restrict__ out) {
  __shared__ float red[256];
  red[threadIdx.x] = partial[threadIdx.x];
  __syncthreads();
  for (int s = 128; s > 0; s >>= 1) {
    if (threadIdx.x < s) red[threadIdx.x] += red[threadIdx.x + s];
    __syncthreads();
  }
  if (threadIdx.x == 0) out[0] = red[0];
}

// ---------------------------------------------------------------------------
extern "C" void kernel_launch(void* const* d_in, const int* in_sizes, int n_in,
                              void* d_out, int out_size, void* d_ws, size_t ws_size,
                              hipStream_t stream) {
  (void)in_sizes; (void)n_in; (void)out_size; (void)ws_size;
  const float* x   = (const float*)d_in[0];   // [256 x 256]
  const float* W1  = (const float*)d_in[1];   // [1024 x 256]
  const float* b1  = (const float*)d_in[2];   // [1024]
  const float* W2  = (const float*)d_in[3];   // [512 x 1024]
  const float* b2  = (const float*)d_in[4];   // [512]
  const float* W3  = (const float*)d_in[5];   // [256 x 512]
  const float* b3  = (const float*)d_in[6];   // [256]
  const int*   kp  = (const int*)d_in[7];     // scalar k

  // workspace carve (256B aligned), total ~10 MB
  char* base = (char*)d_ws;
  size_t off = 0;
  auto carve = [&](size_t bytes) -> void* {
    void* p = base + off;
    off = (off + bytes + 255) & ~(size_t)255;
    return p;
  };
  _Float16* xhi   = (_Float16*)carve(256u * 256u * 2u);
  _Float16* xlo   = (_Float16*)carve(256u * 256u * 2u);
  _Float16* w1hi  = (_Float16*)carve(1024u * 256u * 2u);
  _Float16* w1lo  = (_Float16*)carve(1024u * 256u * 2u);
  _Float16* w1Thi = (_Float16*)carve(1024u * 256u * 2u);  // [256 x 1024]
  _Float16* w1Tlo = (_Float16*)carve(1024u * 256u * 2u);
  _Float16* w2hi  = (_Float16*)carve(512u * 1024u * 2u);
  _Float16* w2lo  = (_Float16*)carve(512u * 1024u * 2u);
  _Float16* w3hi  = (_Float16*)carve(256u * 512u * 2u);
  _Float16* w3lo  = (_Float16*)carve(256u * 512u * 2u);
  float*    w1n   = (float*)carve(1024u * 4u);
  float*    z1    = (float*)carve(256u * 1024u * 4u);
  _Float16* h1hi  = (_Float16*)carve(256u * 1024u * 2u);
  _Float16* h1lo  = (_Float16*)carve(256u * 1024u * 2u);
  unsigned* m1b   = (unsigned*)carve(256u * 32u * 4u);
  float*    z2    = (float*)carve(256u * 512u * 4u);
  _Float16* h2hi  = (_Float16*)carve(256u * 512u * 2u);
  _Float16* h2lo  = (_Float16*)carve(256u * 512u * 2u);
  unsigned* m2b   = (unsigned*)carve(256u * 16u * 4u);
  float*    z3    = (float*)carve(256u * 256u * 4u);
  float*    dists = (float*)carve(256u * 1792u * 4u);
  float*    part  = (float*)carve(256u * 4u);

  // prep: hi/lo plane splits, transpose, W1 row norms
  cvt_split<<<128, 256, 0, stream>>>(x, xhi, xlo, 256 * 256);
  cvt_split<<<512, 256, 0, stream>>>(W1, w1hi, w1lo, 1024 * 256);
  cvt_split<<<1024, 256, 0, stream>>>(W2, w2hi, w2lo, 512 * 1024);
  cvt_split<<<256, 256, 0, stream>>>(W3, w3hi, w3lo, 256 * 512);
  transpose_split<<<512, 256, 0, stream>>>(W1, w1Thi, w1Tlo, 1024, 256);
  rownorm<<<1024, 256, 0, stream>>>(W1, w1n, 256);

  // layer 1: z1 = x @ W1^T + b1  (BT plane == W1 planes directly)
  gemm_bias_store<<<128, 256, 0, stream>>>(xhi, xlo, w1hi, w1lo, b1, z1, 256, 1024, 256);
  pack_layer1<<<1024, 256, 0, stream>>>(z1, w1n, h1hi, h1lo, m1b, dists);

  // layer 2: z2 = h1 @ W2^T + b2
  gemm_bias_store<<<64, 256, 0, stream>>>(h1hi, h1lo, w2hi, w2lo, b2, z2, 256, 512, 1024);
  pack_layer2<<<512, 256, 0, stream>>>(z2, h2hi, h2lo, m2b);

  // layer 3: z3 = h2 @ W3^T + b3
  gemm_bias_store<<<32, 256, 0, stream>>>(h2hi, h2lo, w3hi, w3lo, b3, z3, 256, 256, 512);

  // heavy per-batch V2/V3 norm GEMMs -> dist2, dist3
  border_norms<<<256, 256, 0, stream>>>(w2hi, w2lo, w1Thi, w1Tlo, w3hi, w3lo,
                                        m1b, m2b, z2, z3, dists);

  // k smallest per row, then deterministic total
  topk_partial<<<256, 256, 0, stream>>>(dists, kp, part);
  final_sum<<<1, 256, 0, stream>>>(part, (float*)d_out);
}
